// DiTBlock_49890340110860
// MI455X (gfx1250) — matrix operand order repaired and checked by reading, not compiled, not run on hardware
//
#include <hip/hip_runtime.h>
#include <math.h>

// ---------------- constants ----------------
#define H_   1024
#define NH_  16
#define HD_  64
#define B_   4
#define N_   1024
#define S_   77
#define MLP_ 4096
#define BN_  (B_*N_)

typedef __attribute__((ext_vector_type(16))) __bf16 v16bf;
typedef __attribute__((ext_vector_type(8)))  float  v8f;

union Frag {
  v16bf v;
  unsigned short u[16];
  uint4 q[2];
};

__device__ __forceinline__ unsigned short bf16u(float f) {
  union { __bf16 h; unsigned short u; } c;
  c.h = (__bf16)f;                 // hardware RNE convert (v_cvt_*bf16_f32)
  return c.u;
}
__device__ __forceinline__ unsigned packbf2(float a, float b) {
  return (unsigned)bf16u(a) | ((unsigned)bf16u(b) << 16);
}

// ---------------- conditioning: cproj = silu(c) @ W_mod + b_mod ----------------
__global__ __launch_bounds__(256)
void cond_kernel(const float* __restrict__ c, const float* __restrict__ Wm,
                 const float* __restrict__ bm, float* __restrict__ cproj) {
  __shared__ float sc[H_];
  const int b = blockIdx.y;
  for (int i = threadIdx.x; i < H_; i += 256) {
    float cv = c[b * H_ + i];
    sc[i] = cv / (1.f + __expf(-cv));
  }
  __syncthreads();
  const int j = blockIdx.x * 256 + threadIdx.x;
  const float* wp = Wm + j;
  float acc = 0.f;
  for (int i = 0; i < H_; ++i)
    acc = fmaf(sc[i], wp[i * (6 * H_)], acc);
  cproj[b * (6 * H_) + j] = acc + bm[j];
}

// ---------------- LayerNorm (+ optional adaLN modulate) ----------------
__global__ __launch_bounds__(256)
void ln_kernel(const float* __restrict__ in, float* __restrict__ out,
               const float* __restrict__ cproj, int shift_chunk, int scale_chunk,
               int modulated, int rows_per_batch) {
  __shared__ float red[8];
  const int row = blockIdx.x;
  const float* xr = in + row * H_;
  float* orow = out + row * H_;
  const int tid = threadIdx.x;
  float lx[4]; float s = 0.f;
  #pragma unroll
  for (int i = 0; i < 4; ++i) { lx[i] = xr[tid + i * 256]; s += lx[i]; }
  #pragma unroll
  for (int off = 16; off >= 1; off >>= 1) s += __shfl_xor(s, off, 32);
  if ((tid & 31) == 0) red[tid >> 5] = s;
  __syncthreads();
  float tot = 0.f;
  #pragma unroll
  for (int i = 0; i < 8; ++i) tot += red[i];
  const float mu = tot * (1.f / H_);
  __syncthreads();
  float v = 0.f;
  #pragma unroll
  for (int i = 0; i < 4; ++i) { float d = lx[i] - mu; v += d * d; }
  #pragma unroll
  for (int off = 16; off >= 1; off >>= 1) v += __shfl_xor(v, off, 32);
  if ((tid & 31) == 0) red[tid >> 5] = v;
  __syncthreads();
  float var = 0.f;
  #pragma unroll
  for (int i = 0; i < 8; ++i) var += red[i];
  const float rstd = rsqrtf(var * (1.f / H_) + 1e-6f);
  const int b = row / rows_per_batch;
  #pragma unroll
  for (int i = 0; i < 4; ++i) {
    int j = tid + i * 256;
    float y = (lx[i] - mu) * rstd;
    if (modulated) {
      float sc = cproj[b * 6 * H_ + scale_chunk * H_ + j];
      float sh = cproj[b * 6 * H_ + shift_chunk * H_ + j];
      y = y * (1.f + sc) + sh;
    }
    orow[j] = y;
  }
}

// ---------------- bf16 WMMA GEMM: C = epi(A@W + bias) ----------------
#define TBM 128
#define TBN 128
#define TBK 32
#define LDA 40
#define LDB 40

#define EPI_NONE 0
#define EPI_GELU 1
#define EPI_RES  2

template<int EPI>
__global__ __launch_bounds__(256)
void gemm_bf16_wmma(const float* __restrict__ A, const float* __restrict__ W,
                    const float* __restrict__ bias,
                    const float* __restrict__ res, const float* __restrict__ gate,
                    float* __restrict__ C, int M, int Nn, int K, int rows_per_batch) {
  __shared__ __align__(16) unsigned short sA[TBM * LDA];
  __shared__ __align__(16) unsigned short sB[TBN * LDB];   // transposed: [ncol][k]
  const int tid  = threadIdx.x;
  const int lane = tid & 31;
  const int wid  = tid >> 5;
  const int wm = wid >> 1;        // 0..3 -> 32 rows each
  const int wn = wid & 1;         // 0..1 -> 64 cols each
  const int mb = blockIdx.y * TBM;
  const int nb = blockIdx.x * TBN;
  const int hi  = lane >> 4;
  const int l16 = lane & 15;

  // staging decomposition (32-bit indexing only)
  const int a_r0 = tid >> 3;          // 0..31, +32 per step (4 steps)
  const int a_kk = (tid & 7) * 4;     // float4 column
  const int b_j  = tid & 127;
  const int b_k0 = tid >> 7;          // 0..1, +2 per step (16 steps)

  v8f acc[2][4];
  #pragma unroll
  for (int i = 0; i < 2; ++i)
    #pragma unroll
    for (int j = 0; j < 4; ++j)
      #pragma unroll
      for (int r = 0; r < 8; ++r) acc[i][j][r] = 0.f;

  for (int kb = 0; kb < K; kb += TBK) {
    // stage A: 128x32 f32 -> bf16, float4 loads + ds_write_b64
    #pragma unroll
    for (int i = 0; i < 4; ++i) {
      int r = a_r0 + i * 32;
      int row = mb + r;
      float4 fv = {0.f, 0.f, 0.f, 0.f};
      if (row < M) fv = *(const float4*)(A + row * K + kb + a_kk);
      uint2 p;
      p.x = packbf2(fv.x, fv.y);
      p.y = packbf2(fv.z, fv.w);
      *(uint2*)&sA[r * LDA + a_kk] = p;
    }
    // stage W^T: 32x128 f32 -> bf16 (coalesced global read, scalar LDS write)
    {
      const float* wp = W + nb + b_j;
      #pragma unroll
      for (int i = 0; i < 16; ++i) {
        int kk = b_k0 + i * 2;
        sB[b_j * LDB + kk] = bf16u(wp[(kb + kk) * Nn]);
      }
    }
    __syncthreads();

    Frag af[2], bf[4];
    #pragma unroll
    for (int mt = 0; mt < 2; ++mt) {                       // A frag: 16x32 layout
      int m = wm * 32 + mt * 16 + l16;
      int kbase = hi ? 8 : 0;
      af[mt].q[0] = *(const uint4*)&sA[m * LDA + kbase];
      af[mt].q[1] = *(const uint4*)&sA[m * LDA + 16 + kbase];
    }
    #pragma unroll
    for (int nt = 0; nt < 4; ++nt) {                       // B frag: 32x16 layout
      int j = wn * 64 + nt * 16 + l16;
      int kbase = hi ? 8 : 0;
      bf[nt].q[0] = *(const uint4*)&sB[j * LDB + kbase];
      bf[nt].q[1] = *(const uint4*)&sB[j * LDB + 16 + kbase];
    }
    #pragma unroll
    for (int mt = 0; mt < 2; ++mt)
      #pragma unroll
      for (int nt = 0; nt < 4; ++nt)
        acc[mt][nt] = __builtin_amdgcn_wmma_f32_16x16x32_bf16(
            false, af[mt].v, false, bf[nt].v, (short)0, acc[mt][nt], false, false);
    __syncthreads();
  }

  #pragma unroll
  for (int mt = 0; mt < 2; ++mt)
    #pragma unroll
    for (int r = 0; r < 8; ++r) {
      int row = mb + wm * 32 + mt * 16 + r + hi * 8;
      if (row >= M) continue;
      float* crow = C + row * Nn;
      const float* rrow = (EPI == EPI_RES) ? res + row * Nn : nullptr;
      const float* grow = (EPI == EPI_RES && gate)
                              ? gate + (row / rows_per_batch) * (6 * H_) : nullptr;
      #pragma unroll
      for (int nt = 0; nt < 4; ++nt) {
        int col = nb + wn * 64 + nt * 16 + l16;
        float v = acc[mt][nt][r] + bias[col];
        if (EPI == EPI_GELU) {
          v = 0.5f * v * (1.f + tanhf(0.7978845608028654f * (v + 0.044715f * v * v * v)));
        } else if (EPI == EPI_RES) {
          float g = grow ? grow[col] : 1.f;
          v = rrow[col] + g * v;
        }
        crow[col] = v;
      }
    }
}

// ---------------- flash attention (WMMA QK^T + online softmax + WMMA PV) -----
__global__ __launch_bounds__(256)
void flash_attn(const float* __restrict__ qb, int q_bs, int q_rs,
                const float* __restrict__ kb_, int k_bs, int k_rs,
                const float* __restrict__ vb, int v_bs, int v_rs,
                float* __restrict__ ob, int o_bs, int nk, float scale) {
  __shared__ __align__(16) unsigned short pl[8][16 * 32];
  const int lane = threadIdx.x & 31;
  const int w    = threadIdx.x >> 5;
  const int hi   = lane >> 4;
  const int l16  = lane & 15;
  const int b = blockIdx.z, h = blockIdx.y;
  const int q0 = blockIdx.x * 128 + w * 16;

  const float* qp = qb  + b * q_bs + h * HD_;
  const float* kp = kb_ + b * k_bs + h * HD_;
  const float* vp = vb  + b * v_bs + h * HD_;
  float*       op = ob  + b * o_bs + h * HD_;

  Frag qa[2];                                   // Q (pre-scaled), 16x64 as 2 frags
  {
    const float4* q4 = (const float4*)(qp + (q0 + l16) * q_rs);
    const int kb4 = hi ? 4 : 0;                 // kbase/4
    #pragma unroll
    for (int c2 = 0; c2 < 2; ++c2) {
      float4 lo0 = q4[c2 * 8 + kb4];
      float4 lo1 = q4[c2 * 8 + kb4 + 1];
      float4 hi0 = q4[c2 * 8 + kb4 + 2];
      float4 hi1 = q4[c2 * 8 + kb4 + 3];
      qa[c2].q[0].x = packbf2(scale * lo0.x, scale * lo0.y);
      qa[c2].q[0].y = packbf2(scale * lo0.z, scale * lo0.w);
      qa[c2].q[0].z = packbf2(scale * lo1.x, scale * lo1.y);
      qa[c2].q[0].w = packbf2(scale * lo1.z, scale * lo1.w);
      qa[c2].q[1].x = packbf2(scale * hi0.x, scale * hi0.y);
      qa[c2].q[1].y = packbf2(scale * hi0.z, scale * hi0.w);
      qa[c2].q[1].z = packbf2(scale * hi1.x, scale * hi1.y);
      qa[c2].q[1].w = packbf2(scale * hi1.z, scale * hi1.w);
    }
  }

  v8f o[4];
  #pragma unroll
  for (int t = 0; t < 4; ++t)
    #pragma unroll
    for (int r = 0; r < 8; ++r) o[t][r] = 0.f;
  float mrow[8], lrow[8];
  #pragma unroll
  for (int r = 0; r < 8; ++r) { mrow[r] = -3.0e38f; lrow[r] = 0.f; }

  for (int kb0 = 0; kb0 < nk; kb0 += 32) {
    v8f s2[2];
    #pragma unroll
    for (int h2 = 0; h2 < 2; ++h2) {            // scores, 16 keys per tile
      int key = kb0 + h2 * 16 + l16;
      int kcl = key < nk ? key : nk - 1;
      const float4* k4 = (const float4*)(kp + kcl * k_rs);
      const int db4 = hi ? 4 : 0;               // db/4
      Frag kf0, kf1;
      #pragma unroll
      for (int i = 0; i < 4; ++i) {
        float4 fv = k4[db4 + i];
        ((unsigned*)&kf0)[2 * i]     = packbf2(fv.x, fv.y);
        ((unsigned*)&kf0)[2 * i + 1] = packbf2(fv.z, fv.w);
      }
      #pragma unroll
      for (int i = 0; i < 4; ++i) {
        float4 fv = k4[8 + db4 + i];
        ((unsigned*)&kf1)[2 * i]     = packbf2(fv.x, fv.y);
        ((unsigned*)&kf1)[2 * i + 1] = packbf2(fv.z, fv.w);
      }
      v8f sacc;
      #pragma unroll
      for (int r = 0; r < 8; ++r) sacc[r] = 0.f;
      sacc = __builtin_amdgcn_wmma_f32_16x16x32_bf16(false, qa[0].v, false, kf0.v, (short)0, sacc, false, false);
      sacc = __builtin_amdgcn_wmma_f32_16x16x32_bf16(false, qa[1].v, false, kf1.v, (short)0, sacc, false, false);
      if (key >= nk) {
        #pragma unroll
        for (int r = 0; r < 8; ++r) sacc[r] = -3.0e38f;
      }
      s2[h2] = sacc;
    }
    float alpha[8];
    #pragma unroll
    for (int r = 0; r < 8; ++r) {               // online softmax (16-lane groups)
      float mx = fmaxf(s2[0][r], s2[1][r]);
      #pragma unroll
      for (int off = 8; off >= 1; off >>= 1) mx = fmaxf(mx, __shfl_xor(mx, off, 32));
      float mnew = fmaxf(mrow[r], mx);
      alpha[r] = __expf(mrow[r] - mnew);
      mrow[r] = mnew;
      float p0 = __expf(s2[0][r] - mnew);
      float p1 = __expf(s2[1][r] - mnew);
      s2[0][r] = p0; s2[1][r] = p1;
      float sum = p0 + p1;
      #pragma unroll
      for (int off = 8; off >= 1; off >>= 1) sum += __shfl_xor(sum, off, 32);
      lrow[r] = lrow[r] * alpha[r] + sum;
    }
    #pragma unroll
    for (int t = 0; t < 4; ++t)
      #pragma unroll
      for (int r = 0; r < 8; ++r) o[t][r] *= alpha[r];
    #pragma unroll
    for (int r = 0; r < 8; ++r) {               // P: C-layout -> LDS (bf16)
      int rr = r + hi * 8;
      pl[w][rr * 32 + l16]      = bf16u(s2[0][r]);
      pl[w][rr * 32 + 16 + l16] = bf16u(s2[1][r]);
    }
    asm volatile("s_wait_dscnt 0" ::: "memory"); // wave-local LDS transpose fence
    Frag pa;                                    // P: LDS -> A-layout (16B reads)
    {
      int kbase = hi ? 16 : 0;
      pa.q[0] = *(const uint4*)&pl[w][l16 * 32 + kbase];
      pa.q[1] = *(const uint4*)&pl[w][l16 * 32 + kbase + 8];
    }
    #pragma unroll
    for (int t = 0; t < 4; ++t) {               // PV
      Frag vf;
      const float* vcol = vp + t * 16 + l16;
      const int e0 = hi ? 16 : 0;
      #pragma unroll
      for (int e = 0; e < 16; e += 2) {
        int key0 = kb0 + e0 + e;
        int k0 = key0 < nk ? key0 : nk - 1;
        int k1 = key0 + 1 < nk ? key0 + 1 : nk - 1;
        ((unsigned*)&vf)[e >> 1] = packbf2(vcol[k0 * v_rs], vcol[k1 * v_rs]);
      }
      o[t] = __builtin_amdgcn_wmma_f32_16x16x32_bf16(false, pa.v, false, vf.v, (short)0, o[t], false, false);
    }
    asm volatile("s_wait_dscnt 0" ::: "memory");
  }

  #pragma unroll
  for (int r = 0; r < 8; ++r) {
    float inv = 1.f / lrow[r];
    int n = q0 + r + hi * 8;
    float* orow = op + n * H_;
    #pragma unroll
    for (int t = 0; t < 4; ++t)
      orow[t * 16 + l16] = o[t][r] * inv;
  }
}

// ---------------- host side ----------------
extern "C" void kernel_launch(void* const* d_in, const int* in_sizes, int n_in,
                              void* d_out, int out_size, void* d_ws, size_t ws_size,
                              hipStream_t stream) {
  (void)in_sizes; (void)n_in; (void)out_size; (void)ws_size;
  const float* x      = (const float*)d_in[0];
  const float* c      = (const float*)d_in[1];
  const float* ctx    = (const float*)d_in[2];
  const float* W_mod  = (const float*)d_in[3];
  const float* b_mod  = (const float*)d_in[4];
  const float* W_qkv  = (const float*)d_in[5];
  const float* b_qkv  = (const float*)d_in[6];
  const float* W_proj = (const float*)d_in[7];
  const float* b_proj = (const float*)d_in[8];
  const float* W_qc   = (const float*)d_in[9];
  const float* b_qc   = (const float*)d_in[10];
  const float* W_kvc  = (const float*)d_in[11];
  const float* b_kvc  = (const float*)d_in[12];
  const float* W_cproj= (const float*)d_in[13];
  const float* b_cproj= (const float*)d_in[14];
  const float* W_mlp1 = (const float*)d_in[15];
  const float* b_mlp1 = (const float*)d_in[16];
  const float* W_mlp2 = (const float*)d_in[17];
  const float* b_mlp2 = (const float*)d_in[18];
  float* out = (float*)d_out;

  float* f     = (float*)d_ws;
  float* cproj = f;                              // B*6H
  float* xn    = cproj + (size_t)6 * H_ * B_;    // BN*H
  float* big   = xn + (size_t)BN_ * H_;          // BN*MLP (qkv then h)
  float* tb    = big + (size_t)BN_ * MLP_;       // BN*H (attn_out / cattn_out)
  float* x1    = tb + (size_t)BN_ * H_;          // BN*H running residual
  float* qc    = x1 + (size_t)BN_ * H_;          // BN*H
  float* lnctx = qc + (size_t)BN_ * H_;          // B*S*H
  float* kvc   = lnctx + (size_t)B_ * S_ * H_;   // B*S*2H

  const dim3 blk(256);
  const float scale = 0.125f;                    // 64^-0.5

  cond_kernel<<<dim3(6 * H_ / 256, B_), blk, 0, stream>>>(c, W_mod, b_mod, cproj);

  // ---- self attention ----
  ln_kernel<<<dim3(BN_), blk, 0, stream>>>(x, xn, cproj, 0, 1, 1, N_);
  gemm_bf16_wmma<EPI_NONE><<<dim3(3 * H_ / 128, BN_ / 128), blk, 0, stream>>>(
      xn, W_qkv, b_qkv, nullptr, nullptr, big, BN_, 3 * H_, H_, N_);
  flash_attn<<<dim3(N_ / 128, NH_, B_), blk, 0, stream>>>(
      big,          N_ * 3 * H_, 3 * H_,
      big + H_,     N_ * 3 * H_, 3 * H_,
      big + 2 * H_, N_ * 3 * H_, 3 * H_,
      tb,           N_ * H_, N_, scale);
  gemm_bf16_wmma<EPI_RES><<<dim3(H_ / 128, BN_ / 128), blk, 0, stream>>>(
      tb, W_proj, b_proj, x, cproj + 2 * H_, x1, BN_, H_, H_, N_);

  // ---- cross attention ----
  ln_kernel<<<dim3(BN_), blk, 0, stream>>>(x1, xn, nullptr, 0, 0, 0, N_);
  gemm_bf16_wmma<EPI_NONE><<<dim3(H_ / 128, BN_ / 128), blk, 0, stream>>>(
      xn, W_qc, b_qc, nullptr, nullptr, qc, BN_, H_, H_, N_);
  ln_kernel<<<dim3(B_ * S_), blk, 0, stream>>>(ctx, lnctx, nullptr, 0, 0, 0, S_);
  gemm_bf16_wmma<EPI_NONE><<<dim3(2 * H_ / 128, (B_ * S_ + 127) / 128), blk, 0, stream>>>(
      lnctx, W_kvc, b_kvc, nullptr, nullptr, kvc, B_ * S_, 2 * H_, H_, S_);
  flash_attn<<<dim3(N_ / 128, NH_, B_), blk, 0, stream>>>(
      qc,        N_ * H_, H_,
      kvc,       S_ * 2 * H_, 2 * H_,
      kvc + H_,  S_ * 2 * H_, 2 * H_,
      tb,        N_ * H_, S_, scale);
  gemm_bf16_wmma<EPI_RES><<<dim3(H_ / 128, BN_ / 128), blk, 0, stream>>>(
      tb, W_cproj, b_cproj, x1, nullptr, x1, BN_, H_, H_, N_);

  // ---- MLP ----
  ln_kernel<<<dim3(BN_), blk, 0, stream>>>(x1, xn, cproj, 3, 4, 1, N_);
  gemm_bf16_wmma<EPI_GELU><<<dim3(MLP_ / 128, BN_ / 128), blk, 0, stream>>>(
      xn, W_mlp1, b_mlp1, nullptr, nullptr, big, BN_, MLP_, H_, N_);
  gemm_bf16_wmma<EPI_RES><<<dim3(H_ / 128, BN_ / 128), blk, 0, stream>>>(
      big, W_mlp2, b_mlp2, x1, cproj + 5 * H_, out, BN_, H_, MLP_, N_);
}
